// TheNetwork_25623774888596
// MI455X (gfx1250) — matrix operand order repaired and checked by reading, not compiled, not run on hardware
//
#include <hip/hip_runtime.h>
#include <hip/hip_bf16.h>

#define H    256
#define TVL  2048
#define TQL  128
#define BN   160      // B * NC
#define NCH  5
#define NSV  4        // video sequence split (parallelism): 640 blocks, 512 rows each
#define SAS  264      // padded LDS row stride (halves) for the A tile

typedef __attribute__((ext_vector_type(16))) __bf16         v16bf;
typedef __attribute__((ext_vector_type(16))) unsigned short v16u;
typedef __attribute__((ext_vector_type(8)))  float          v8f;

__device__ __forceinline__ unsigned short f2bf(float f) {
    unsigned int u = __float_as_uint(f);
    u += 0x7FFFu + ((u >> 16) & 1u);          // round-to-nearest-even
    return (unsigned short)(u >> 16);
}
__device__ __forceinline__ float bf2f(unsigned short s) {
    return __uint_as_float(((unsigned int)s) << 16);
}
// CDNA5 hardware transcendentals. ISA 7.4: one independent op / V_NOP after a
// TRANS op before its output is consumed -> bake the v_nop into the asm.
__device__ __forceinline__ float fast_tanh(float x) {
    float r;
    asm volatile("v_tanh_f32 %0, %1\n\tv_nop" : "=v"(r) : "v"(x));
    return r;
}
__device__ __forceinline__ float fast_exp2(float x) {
    float r;
    asm volatile("v_exp_f32 %0, %1\n\tv_nop" : "=v"(r) : "v"(x));
    return r;
}
__device__ __forceinline__ float fast_exp(float x) {
    return fast_exp2(x * 1.4426950408889634f);
}

// ---------------------------------------------------------------------------
// One-time: Wav/Wat f32 [k][n] -> bf16 transposed [n][k] in workspace.
// ---------------------------------------------------------------------------
__global__ __launch_bounds__(256) void transw_kernel(const float* __restrict__ Wav,
                                                     const float* __restrict__ Wat,
                                                     unsigned short* __restrict__ WavT,
                                                     unsigned short* __restrict__ WatT) {
    const float* W = blockIdx.x ? Wat : Wav;
    unsigned short* WT = blockIdx.x ? WatT : WavT;
    for (int e = threadIdx.x; e < H * H; e += 256) {
        const int k = e >> 8, n = e & 255;
        WT[n * H + k] = f2bf(W[e]);
    }
}

// ---------------------------------------------------------------------------
// Small GEMVs: uv = hh@Uav + bav, ut = hh@Uat + bat, wbs = hh@Wb
// ---------------------------------------------------------------------------
__global__ __launch_bounds__(256) void prep_kernel(const float* __restrict__ h,
                                                   const float* __restrict__ Uav,
                                                   const float* __restrict__ Uat,
                                                   const float* __restrict__ Wb,
                                                   const float* __restrict__ bav,
                                                   const float* __restrict__ bat,
                                                   float* __restrict__ uv,
                                                   float* __restrict__ ut,
                                                   float* __restrict__ wbs) {
    __shared__ float sh[H];
    const int bn = blockIdx.x, t = threadIdx.x;
    sh[t] = h[bn * H + t];
    __syncthreads();
    float a0 = 0.f, a1 = 0.f, a2 = 0.f;
    for (int k = 0; k < H; ++k) {
        const float x = sh[k];
        a0 += x * Uav[k * H + t];
        a1 += x * Uat[k * H + t];
        a2 += x * Wb [k * H + t];
    }
    uv [bn * H + t] = a0 + bav[t];
    ut [bn * H + t] = a1 + bat[t];
    wbs[bn * H + t] = a2;
}

// ---------------------------------------------------------------------------
// Fused attention over a slice of X[bn] (rows [part*rpp, (part+1)*rpp)):
//   logits = tanh(X@W + u) . V       (bf16 WMMA, V_TANH_F32 epilogue)
//   online masked softmax, partial (m, Z, unnormalized acc) written out.
// ---------------------------------------------------------------------------
__global__ __launch_bounds__(256) void attn_kernel(const float* __restrict__ X,
                                                   const unsigned short* __restrict__ WT,
                                                   const float* __restrict__ uvec,
                                                   const float* __restrict__ Vvec,
                                                   const int* __restrict__ lens,
                                                   int is_video, int Trows, int nsplit,
                                                   float* __restrict__ pm,
                                                   float* __restrict__ pz,
                                                   float* __restrict__ pacc) {
    extern __shared__ char smem[];
    unsigned short* sW   = (unsigned short*)smem;                       // 256*256 bf16 (W^T)
    unsigned short* sA   = (unsigned short*)(smem + H * H * 2);         // 128*SAS bf16 tile
    float* sV     = (float*)(smem + H * H * 2 + 128 * SAS * 2);
    float* sU     = sV + H;
    float* sEv    = sU + H;
    float* sWgt   = sEv + 128;
    float* sRed   = sWgt + 128;
    float* sState = sRed + 8;     // [0]=running max, [1]=running Z, [2]=rescale

    const int bn    = blockIdx.x / nsplit;
    const int part  = blockIdx.x % nsplit;
    const int rpp   = Trows / nsplit;
    const int tid   = threadIdx.x;
    const int lane  = tid & 31;
    const int wave  = tid >> 5;
    const int nlane = lane & 15;

    {   // W^T copy as dwords (coalesced)
        const unsigned int* src = (const unsigned int*)WT;
        unsigned int* dst = (unsigned int*)sW;
        for (int e = tid; e < H * H / 2; e += 256) dst[e] = src[e];
    }
    sU[tid] = uvec[bn * H + tid];
    sV[tid] = Vvec[tid];
    if (tid == 0) { sState[0] = -3.0e38f; sState[1] = 0.f; sState[2] = 0.f; }

    int limit;
    if (is_video) {
        limit = lens[bn / NCH];
    } else {
        const int b = bn / NCH;
        limit = lens[b * NCH];
        for (int j = 1; j < NCH; ++j) { const int v = lens[b * NCH + j]; limit = v > limit ? v : limit; }
    }
    __syncthreads();

    float accv = 0.f;
    const int ntiles = rpp >> 7;
    for (int tile = 0; tile < ntiles; ++tile) {
        const int rowbase = part * rpp + tile * 128;
        // ---- stage 1: global f32 -> LDS bf16 tile [128][SAS] ----
        const float* gX = X + ((size_t)bn * Trows + (size_t)rowbase) * H;
        #pragma unroll
        for (int i = 0; i < 32; ++i) {
            const int e = i * 1024 + tid * 4;
            const float4 v = *(const float4*)(gX + e);
            const int row = e >> 8, col = e & 255;
            unsigned short* p = &sA[row * SAS + col];
            p[0] = f2bf(v.x); p[1] = f2bf(v.y); p[2] = f2bf(v.z); p[3] = f2bf(v.w);
        }
        __syncthreads();

        // ---- stage 2: WMMA GEMM + tanh().V epilogue -> per-row logits ----
        float rowpart[8];
        #pragma unroll
        for (int r = 0; r < 8; ++r) rowpart[r] = 0.f;
        const int arow  = (wave << 4) + nlane;
        const int kAofs = (lane >> 4) << 4;
        const int kBofs = (lane >> 4) << 4;
        for (int nt = 0; nt < 16; ++nt) {
            v8f acc = {};
            #pragma unroll
            for (int kt = 0; kt < 8; ++kt) {
                const unsigned short* pa = &sA[arow * SAS + kt * 32 + kAofs];
                v16u ua;
                #pragma unroll
                for (int j = 0; j < 8; ++j) ua[j] = pa[j];
                #pragma unroll
                for (int j = 0; j < 8; ++j) ua[8 + j] = pa[8 + j];
                const unsigned short* pb = &sW[(nt * 16 + nlane) * H + kt * 32 + kBofs];
                v16u ub;
                #pragma unroll
                for (int j = 0; j < 16; ++j) ub[j] = pb[j];
                acc = __builtin_amdgcn_wmma_f32_16x16x32_bf16(
                        false, __builtin_bit_cast(v16bf, ua),
                        false, __builtin_bit_cast(v16bf, ub),
                        (short)0, acc, false, false);
            }
            const float uvh = sU[nt * 16 + nlane];
            const float vah = sV[nt * 16 + nlane];
            #pragma unroll
            for (int r = 0; r < 8; ++r)
                rowpart[r] += fast_tanh(acc[r] + uvh) * vah;
        }
        #pragma unroll
        for (int r = 0; r < 8; ++r) {
            float v = rowpart[r];
            v += __shfl_xor(v, 1, 32);
            v += __shfl_xor(v, 2, 32);
            v += __shfl_xor(v, 4, 32);
            v += __shfl_xor(v, 8, 32);
            if (nlane == 0) {
                const int row = (wave << 4) + ((lane >> 4) << 3) + r;
                const int t   = rowbase + row;
                sEv[row] = (t < limit) ? v : -1.0e30f;
            }
        }
        __syncthreads();

        // ---- stage 3: online softmax (running max / running Z) ----
        float lm = (tid < 128) ? sEv[tid] : -3.0e38f;
        lm = fmaxf(lm, __shfl_xor(lm, 1, 32));
        lm = fmaxf(lm, __shfl_xor(lm, 2, 32));
        lm = fmaxf(lm, __shfl_xor(lm, 4, 32));
        lm = fmaxf(lm, __shfl_xor(lm, 8, 32));
        lm = fmaxf(lm, __shfl_xor(lm, 16, 32));
        if (lane == 0) sRed[wave] = lm;
        __syncthreads();
        if (tid == 0) {
            float tm = sRed[0];
            for (int w2 = 1; w2 < 8; ++w2) tm = fmaxf(tm, sRed[w2]);
            const float m  = sState[0];
            const float nm = fmaxf(m, tm);
            sState[2] = fast_exp(m - nm);
            sState[0] = nm;
        }
        __syncthreads();
        const float nm = sState[0];
        float w = 0.f;
        if (tid < 128) { w = fast_exp(sEv[tid] - nm); sWgt[tid] = w; }
        float wsum = w;
        wsum += __shfl_xor(wsum, 1, 32);
        wsum += __shfl_xor(wsum, 2, 32);
        wsum += __shfl_xor(wsum, 4, 32);
        wsum += __shfl_xor(wsum, 8, 32);
        wsum += __shfl_xor(wsum, 16, 32);
        if (lane == 0) sRed[wave] = wsum;
        __syncthreads();
        if (tid == 0) {
            float s = 0.f;
            for (int w2 = 0; w2 < 8; ++w2) s += sRed[w2];
            sState[1] = sState[1] * sState[2] + s;
        }
        // ---- stage 4: rescale + accumulate weighted rows (column h = tid) ----
        const float scale = sState[2];
        float s = 0.f;
        for (int t2 = 0; t2 < 128; ++t2)
            s += sWgt[t2] * bf2f(sA[t2 * SAS + tid]);
        accv = accv * scale + s;
        __syncthreads();
    }
    pacc[(size_t)blockIdx.x * H + tid] = accv;
    if (tid == 0) { pm[blockIdx.x] = sState[0]; pz[blockIdx.x] = sState[1]; }
}

// ---------------------------------------------------------------------------
// Merge split partial softmax states: out = sum(acc_p e^{m_p-m}) / sum(Z_p e^{m_p-m})
// ---------------------------------------------------------------------------
__global__ __launch_bounds__(256) void merge_kernel(const float* __restrict__ pm,
                                                    const float* __restrict__ pz,
                                                    const float* __restrict__ pacc,
                                                    int parts, float* __restrict__ out) {
    const int bn = blockIdx.x, t = threadIdx.x;
    float gm = -3.0e38f;
    for (int p = 0; p < parts; ++p) gm = fmaxf(gm, pm[bn * parts + p]);
    float z = 0.f, a = 0.f;
    for (int p = 0; p < parts; ++p) {
        const float s = fast_exp(pm[bn * parts + p] - gm);
        z += pz[bn * parts + p] * s;
        a += pacc[(size_t)(bn * parts + p) * H + t] * s;
    }
    out[bn * H + t] = a / z;
}

// ---------------------------------------------------------------------------
// Final combine: 5 small GEMVs + beta softmax + output tanh.
// ---------------------------------------------------------------------------
__global__ __launch_bounds__(256) void final_kernel(const float* __restrict__ h,
                                                    const float* __restrict__ hv,
                                                    const float* __restrict__ hts,
                                                    const float* __restrict__ Whh,
                                                    const float* __restrict__ bh,
                                                    const float* __restrict__ Wvs,
                                                    const float* __restrict__ bvs,
                                                    const float* __restrict__ Wqs,
                                                    const float* __restrict__ bqs,
                                                    const float* __restrict__ Vbv,
                                                    const float* __restrict__ Vbt,
                                                    const float* __restrict__ bbv,
                                                    const float* __restrict__ bbt,
                                                    const float* __restrict__ wb,
                                                    const float* __restrict__ wbs,
                                                    float* __restrict__ out) {
    __shared__ float shh[H], shv[H], sht[H], sred[16];
    const int bn = blockIdx.x, t = threadIdx.x;
    const int lane = t & 31, wave = t >> 5;
    shh[t] = h  [bn * H + t];
    shv[t] = hv [bn * H + t];
    sht[t] = hts[bn * H + t];
    __syncthreads();
    float dv = 0.f, dt = 0.f, mv = 0.f, mt = 0.f, dh = 0.f;
    for (int k = 0; k < H; ++k) {
        const float xv = shv[k], xt = sht[k], xh = shh[k];
        dv += xv * Wvs[k * H + t];
        dt += xt * Wqs[k * H + t];
        mv += xv * Vbv[k * H + t];
        mt += xt * Vbt[k * H + t];
        dh += xh * Whh[k * H + t];
    }
    const float wg = wbs[bn * H + t];
    float pv = fast_tanh(mv + bbv[t] + wg) * wb[t];
    float pt = fast_tanh(mt + bbt[t] + wg) * wb[t];
    pv += __shfl_xor(pv, 1, 32);  pv += __shfl_xor(pv, 2, 32);
    pv += __shfl_xor(pv, 4, 32);  pv += __shfl_xor(pv, 8, 32);  pv += __shfl_xor(pv, 16, 32);
    pt += __shfl_xor(pt, 1, 32);  pt += __shfl_xor(pt, 2, 32);
    pt += __shfl_xor(pt, 4, 32);  pt += __shfl_xor(pt, 8, 32);  pt += __shfl_xor(pt, 16, 32);
    if (lane == 0) { sred[wave] = pv; sred[8 + wave] = pt; }
    __syncthreads();
    float sv = 0.f, st = 0.f;
    for (int w2 = 0; w2 < 8; ++w2) { sv += sred[w2]; st += sred[8 + w2]; }
    const float mx = fmaxf(sv, st);
    const float e0 = fast_exp(sv - mx), e1 = fast_exp(st - mx);
    const float b0 = e0 / (e0 + e1), b1 = e1 / (e0 + e1);
    out[bn * H + t] = fast_tanh(dh + bh[t] + b0 * (dv + bvs[t]) + b1 * (dt + bqs[t]));
}

// ---------------------------------------------------------------------------
extern "C" void kernel_launch(void* const* d_in, const int* in_sizes, int n_in,
                              void* d_out, int out_size, void* d_ws, size_t ws_size,
                              hipStream_t stream) {
    (void)in_sizes; (void)n_in; (void)out_size; (void)ws_size;
    const float* h   = (const float*)d_in[0];
    const float* hf  = (const float*)d_in[1];
    const float* htx = (const float*)d_in[2];
    const float* Wav = (const float*)d_in[3];
    const float* Wat = (const float*)d_in[4];
    const float* Uav = (const float*)d_in[5];
    const float* Uat = (const float*)d_in[6];
    const float* Vav = (const float*)d_in[7];
    const float* Vat = (const float*)d_in[8];
    const float* bav = (const float*)d_in[9];
    const float* bat = (const float*)d_in[10];
    const float* Whh = (const float*)d_in[11];
    const float* bh  = (const float*)d_in[12];
    const float* Wvs = (const float*)d_in[13];
    const float* bvs = (const float*)d_in[14];
    const float* Wqs = (const float*)d_in[15];
    const float* bqs = (const float*)d_in[16];
    const float* Wb  = (const float*)d_in[17];
    const float* Vbv = (const float*)d_in[18];
    const float* Vbt = (const float*)d_in[19];
    const float* bbv = (const float*)d_in[20];
    const float* bbt = (const float*)d_in[21];
    const float* wb  = (const float*)d_in[22];
    const int* video_lens = (const int*)d_in[23];
    const int* qa_lens    = (const int*)d_in[24];

    float* ws_f = (float*)d_ws;
    float* uv     = ws_f;                  // [BN,H]
    float* ut     = uv     + BN * H;       // [BN,H]
    float* wbs    = ut     + BN * H;       // [BN,H]
    float* hv     = wbs    + BN * H;       // [BN,H]
    float* hts    = hv     + BN * H;       // [BN,H]
    float* pacc_v = hts    + BN * H;       // [BN*NSV,H]
    float* pacc_t = pacc_v + BN * NSV * H; // [BN,H]
    float* pm_v   = pacc_t + BN * H;       // [BN*NSV]
    float* pz_v   = pm_v   + BN * NSV;
    float* pm_t   = pz_v   + BN * NSV;     // [BN]
    float* pz_t   = pm_t   + BN;
    unsigned short* WavT = (unsigned short*)(pz_t + BN);   // [H,H] bf16
    unsigned short* WatT = WavT + H * H;

    transw_kernel<<<2, 256, 0, stream>>>(Wav, Wat, WavT, WatT);
    prep_kernel  <<<BN, 256, 0, stream>>>(h, Uav, Uat, Wb, bav, bat, uv, ut, wbs);

    const size_t smem = (size_t)H * H * 2 + 128 * SAS * 2
                      + (H + H + 128 + 128 + 8 + 4) * sizeof(float);
    attn_kernel<<<BN * NSV, 256, smem, stream>>>(hf,  WavT, uv, Vav, video_lens, 1, TVL, NSV,
                                                 pm_v, pz_v, pacc_v);
    attn_kernel<<<BN,       256, smem, stream>>>(htx, WatT, ut, Vat, qa_lens,    0, TQL, 1,
                                                 pm_t, pz_t, pacc_t);
    merge_kernel<<<BN, 256, 0, stream>>>(pm_v, pz_v, pacc_v, NSV, hv);
    merge_kernel<<<BN, 256, 0, stream>>>(pm_t, pz_t, pacc_t, 1,   hts);

    final_kernel<<<BN, 256, 0, stream>>>(h, hv, hts, Whh, bh, Wvs, bvs, Wqs, bqs,
                                         Vbv, Vbt, bbv, bbt, wb, wbs, (float*)d_out);
}
